// Mlp_30923764532142
// MI455X (gfx1250) — 8-bit matrix operand order repaired and checked by reading, compiled, not run on hardware
//
#include <hip/hip_runtime.h>
#include <cstdint>
#include <cstddef>

typedef int v8i __attribute__((ext_vector_type(8)));
typedef int v4i __attribute__((ext_vector_type(4)));
typedef int v2i __attribute__((ext_vector_type(2)));

#define QP_F 127.0f
#define MIN_RANGE_F 1e-6f

__global__ void init_zero_u32(unsigned* p, int n) {
    int i = blockIdx.x * blockDim.x + threadIdx.x;
    if (i < n) p[i] = 0u;
}

__global__ __launch_bounds__(256) void quant_rows(const float* __restrict__ x, int K,
                                                  int8_t* __restrict__ q,
                                                  float* __restrict__ srow) {
    __shared__ float red[256];
    const int row = blockIdx.x;
    const float* xr = x + (size_t)row * K;
    float m = 0.f;
    for (int j = threadIdx.x; j < K; j += 256) m = fmaxf(m, fabsf(xr[j]));
    red[threadIdx.x] = m;
    __syncthreads();
    for (int s = 128; s > 0; s >>= 1) {
        if (threadIdx.x < s) red[threadIdx.x] = fmaxf(red[threadIdx.x], red[threadIdx.x + s]);
        __syncthreads();
    }
    const float s = fmaxf(red[0], MIN_RANGE_F) / QP_F;
    if (threadIdx.x == 0) srow[row] = s;
    const float inv = 1.0f / s;
    int8_t* qr = q + (size_t)row * K;
    for (int j = threadIdx.x; j < K; j += 256) qr[j] = (int8_t)(int)rintf(xr[j] * inv);
}

__global__ __launch_bounds__(256) void amax_reduce(const float* __restrict__ w, int n,
                                                   unsigned* __restrict__ out) {
    __shared__ float red[256];
    float m = 0.f;
    for (int i = blockIdx.x * blockDim.x + threadIdx.x; i < n; i += gridDim.x * blockDim.x)
        m = fmaxf(m, fabsf(w[i]));
    red[threadIdx.x] = m;
    __syncthreads();
    for (int s = 128; s > 0; s >>= 1) {
        if (threadIdx.x < s) red[threadIdx.x] = fmaxf(red[threadIdx.x], red[threadIdx.x + s]);
        __syncthreads();
    }
    if (threadIdx.x == 0) atomicMax(out, __float_as_uint(red[0]));
}

__global__ __launch_bounds__(256) void quant_tensor(const float* __restrict__ w, int n,
                                                    const unsigned* __restrict__ amax,
                                                    int8_t* __restrict__ q) {
    const float s = fmaxf(__uint_as_float(*amax), MIN_RANGE_F) / QP_F;
    const float inv = 1.0f / s;
    for (int i = blockIdx.x * blockDim.x + threadIdx.x; i < n; i += gridDim.x * blockDim.x)
        q[i] = (int8_t)(int)rintf(w[i] * inv);
}

template <bool GELU>
__global__ __launch_bounds__(256) void gemm_q8(const int8_t* __restrict__ A,
                                               const int8_t* __restrict__ Bm,
                                               const float* __restrict__ bias,
                                               const float* __restrict__ sa,
                                               const unsigned* __restrict__ wamax,
                                               float* __restrict__ out,
                                               int N, int M, int K) {
    const int lane = threadIdx.x & 31;
    const int wave = threadIdx.x >> 5;
    const int hi = lane >> 4;
    const int ln = lane & 15;
    const int t0 = blockIdx.y * 64 + (wave >> 2) * 32;
    const int j0 = blockIdx.x * 128 + (wave & 3) * 32;

    v8i acc[2][2] = {};
    const int nk = K >> 6;
    for (int kk = 0; kk < nk; ++kk) {
        const int k = kk << 6;
        v8i a[2], b[2];
#pragma unroll
        for (int t = 0; t < 2; ++t) {
            const int8_t* ap = A + (size_t)(t0 + t * 16 + ln) * K + k + hi * 16;
            __builtin_prefetch(ap + 64, 0, 0);
#pragma unroll
            for (int c = 0; c < 4; ++c) {
                v2i u = *(const v2i*)(ap + (c >> 1) * 32 + (c & 1) * 8);
                a[t][2 * c] = u[0];
                a[t][2 * c + 1] = u[1];
            }
        }
#pragma unroll
        for (int t = 0; t < 2; ++t) {
            const int8_t* bp = Bm + (size_t)(j0 + t * 16 + ln) * K + k + hi * 16;
            __builtin_prefetch(bp + 64, 0, 0);
            v4i u0 = *(const v4i*)(bp);
            v4i u1 = *(const v4i*)(bp + 32);
#pragma unroll
            for (int c = 0; c < 4; ++c) {
                b[t][c] = u0[c];
                b[t][4 + c] = u1[c];
            }
        }
        acc[0][0] = __builtin_amdgcn_wmma_i32_16x16x64_iu8(true, a[0], true, b[0], acc[0][0], false, false);
        acc[0][1] = __builtin_amdgcn_wmma_i32_16x16x64_iu8(true, a[0], true, b[1], acc[0][1], false, false);
        acc[1][0] = __builtin_amdgcn_wmma_i32_16x16x64_iu8(true, a[1], true, b[0], acc[1][0], false, false);
        acc[1][1] = __builtin_amdgcn_wmma_i32_16x16x64_iu8(true, a[1], true, b[1], acc[1][1], false, false);
    }

    const float sw = fmaxf(__uint_as_float(*wamax), MIN_RANGE_F) / QP_F;
#pragma unroll
    for (int ta = 0; ta < 2; ++ta) {
#pragma unroll
        for (int tb = 0; tb < 2; ++tb) {
            const int j = j0 + tb * 16 + ln;
            const float bj = bias[j];
#pragma unroll
            for (int v = 0; v < 8; ++v) {
                const int i = t0 + ta * 16 + v + hi * 8;
                float val = ((float)acc[ta][tb][v] + bj) * (sa[i] * sw);
                if (GELU) val = 0.5f * val * (1.0f + erff(val * 0.70710678118654752f));
                out[(size_t)i * M + j] = val;
            }
        }
    }
}

extern "C" void kernel_launch(void* const* d_in, const int* in_sizes, int n_in,
                              void* d_out, int out_size, void* d_ws, size_t ws_size,
                              hipStream_t stream) {
    (void)n_in; (void)out_size; (void)ws_size;
    const float* x  = (const float*)d_in[0];
    const float* w1 = (const float*)d_in[1];
    const float* b1 = (const float*)d_in[2];
    const float* w2 = (const float*)d_in[3];
    const float* b2 = (const float*)d_in[4];

    const int H = in_sizes[2];
    const int D = in_sizes[4];
    const int Ntok = in_sizes[0] / D;

    uint8_t* ws = (uint8_t*)d_ws;
    size_t off = 0;
    auto alloc = [&](size_t bytes) -> void* {
        void* p = ws + off;
        off = (off + bytes + 255) & ~(size_t)255;
        return p;
    };
    int8_t*   qx    = (int8_t*)  alloc((size_t)Ntok * D);
    float*    sx    = (float*)   alloc((size_t)Ntok * sizeof(float));
    int8_t*   qw1   = (int8_t*)  alloc((size_t)H * D);
    int8_t*   qw2   = (int8_t*)  alloc((size_t)H * D);
    unsigned* wamax = (unsigned*)alloc(256);
    int8_t*   qh    = (int8_t*)  alloc((size_t)Ntok * H);
    float*    sh    = (float*)   alloc((size_t)Ntok * sizeof(float));
    float*    h     = (float*)   alloc((size_t)Ntok * H * sizeof(float));

    init_zero_u32<<<1, 32, 0, stream>>>(wamax, 2);

    quant_rows<<<Ntok, 256, 0, stream>>>(x, D, qx, sx);

    amax_reduce<<<512, 256, 0, stream>>>(w1, H * D, wamax + 0);
    amax_reduce<<<512, 256, 0, stream>>>(w2, H * D, wamax + 1);
    quant_tensor<<<1024, 256, 0, stream>>>(w1, H * D, wamax + 0, qw1);
    quant_tensor<<<1024, 256, 0, stream>>>(w2, H * D, wamax + 1, qw2);

    gemm_q8<true><<<dim3(H / 128, Ntok / 64), 256, 0, stream>>>(qx, qw1, b1, sx, wamax + 0, h,
                                                                Ntok, H, D);

    quant_rows<<<Ntok, 256, 0, stream>>>(h, H, qh, sh);

    gemm_q8<false><<<dim3(D / 128, Ntok / 64), 256, 0, stream>>>(qh, qw2, b2, sh, wamax + 1,
                                                                 (float*)d_out, Ntok, D, H);
}
